// EvoXMixing_77713138253956
// MI455X (gfx1250) — matrix operand order repaired and checked by reading, not compiled, not run on hardware
//
#include <hip/hip_runtime.h>
#include <hip/hip_bf16.h>

// CDNA5 / gfx1250 WMMA-based batched FWHT + phase rotation + FWHT.
// H_{2^20} = H_1024 (x) H_1024 ; H_1024 = H_32 (x) H_32.
// Each length-1024 transform = two 32x32 Hadamard GEMMs = 8 v_wmma_f32_16x16x32_bf16.
// Pass 2 uses 16-column slabs loaded by TDM with pad_enable (DMA-inserted LDS
// padding: row stride 20 f32 -> b128-aligned, 2-way-max bank conflicts), and
// async LDS->global b128 writeback. ~221 KB LDS / WGP (CDNA5 has 320 KB).

typedef __bf16 v16bf __attribute__((ext_vector_type(16)));
typedef __bf16 v8bf  __attribute__((ext_vector_type(8)));
typedef float  v8f   __attribute__((ext_vector_type(8)));
typedef int    v4i   __attribute__((ext_vector_type(4)));
typedef unsigned short v16u16 __attribute__((ext_vector_type(16)));
typedef unsigned u32x4 __attribute__((ext_vector_type(4)));
typedef int      i32x8 __attribute__((ext_vector_type(8)));
typedef int      i32x4 __attribute__((ext_vector_type(4)));

#define DIMX   1048576   // 2^20
#define BATCHX 16
#define RBS    36        // f32 column stride of rb (padded: bank spread, 16B aligned)
#define STS    40        // bf16 row stride of st (padded, 16B aligned)
#define NCOL   16        // pass-2 slab width (columns)
#define SLS    20        // slab row stride in f32 = 16 data + 4 pad (TDM pad_enable)

#if defined(__gfx1250__) && __has_builtin(__builtin_amdgcn_tensor_load_to_lds)
#define HAVE_TDM 1
#endif
#if defined(__gfx1250__) && __has_builtin(__builtin_amdgcn_global_load_async_to_lds_b128)
#define ASYNC_LOAD 1
#endif
#if defined(__gfx1250__) && __has_builtin(__builtin_amdgcn_global_store_async_from_lds_b128)
#define ASYNC_STORE 1
#endif

#if defined(ASYNC_LOAD) || defined(ASYNC_STORE)
// Exact pointer types per probe: params are v4i* in AS1 (global) / AS3 (LDS).
#define ASYNC_GPTR(p) ((__attribute__((address_space(1))) v4i*)(p))
#define ASYNC_LPTR(p) ((__attribute__((address_space(3))) v4i*)(p))
__device__ __forceinline__ void wait_async0() {
#if __has_builtin(__builtin_amdgcn_s_wait_asynccnt)
  __builtin_amdgcn_s_wait_asynccnt(0);
#else
  asm volatile("s_wait_asynccnt 0x0" ::: "memory");
#endif
}
#endif

#ifdef HAVE_TDM
__device__ __forceinline__ unsigned lds_offset_of(const void* p) {
  // ptrtoint of an addrspace(3) pointer yields the LDS byte offset.
  return (unsigned)(__SIZE_TYPE__)(__attribute__((address_space(3))) const void*)p;
}

// TDM load of a 2D tile: NCOL f32 wide x 1024 rows, global row stride 1024 f32,
// with pad_enable: after every 16 DWORDs insert 4 DWORDs -> LDS row stride SLS.
// D# per CDNA5 ISA ch.8. Groups 2/3 zero (2D tensor).
__device__ __forceinline__ void tdm_load_slab(const float* gptr, unsigned laddr) {
  const unsigned long long ga = (unsigned long long)(__SIZE_TYPE__)gptr;
  u32x4 g0;
  g0[0] = 1u;                                               // count=1 (valid user D#)
  g0[1] = laddr;                                            // lds_addr [63:32]
  g0[2] = (unsigned)(ga & 0xFFFFFFFFu);                     // global_addr[31:0]
  g0[3] = (unsigned)((ga >> 32) & 0x1FFFFFFu) | (2u << 30); // global_addr[56:32]|type=2
  i32x8 g1;
  // [17:16] data_size=2 (4B); [20] pad_enable=1; [24:22] pad_interval=3 (16 DW);
  // [31:25] pad_amount=3 (4 DW)
  g1[0] = (int)0x06D20000;
  g1[1] = (int)((unsigned)NCOL << 16);  // [63:48] tensor_dim0.lo16
  g1[2] = (int)(1024u << 16);           // tensor_dim0.hi=0 ; [95:80] tensor_dim1.lo16
  g1[3] = (int)((unsigned)NCOL << 16);  // tensor_dim1.hi=0 ; [127:112] tile_dim0
  g1[4] = 1024;                         // [143:128] tile_dim1 ; tile_dim2=0
  g1[5] = 1024;                         // tensor_dim0_stride[31:0]
  g1[6] = 0;
  g1[7] = 0;
  const i32x4 z4 = {0, 0, 0, 0};
#if defined(__clang_major__) && __clang_major__ >= 23
  const i32x8 z8 = {0, 0, 0, 0, 0, 0, 0, 0};
  __builtin_amdgcn_tensor_load_to_lds(g0, g1, z4, z4, z8, 0);
#else
  __builtin_amdgcn_tensor_load_to_lds(g0, g1, z4, z4, 0);
#endif
}
#endif  // HAVE_TDM

// Hoisted Hadamard +-1 operand vectors (bf16 +1 = 0x3F80, -1 = 0xBF80).
__device__ __forceinline__ void had_patterns(int lane, v16bf& pa0, v16bf& pa1,
                                             v16bf& pb0, v16bf& pb1) {
  const int half = lane >> 4;
  const int l16  = lane & 15;
  const int kb   = half * 16;
  int ks[16];
#pragma unroll
  for (int i = 0; i < 8; ++i) {
    ks[i]     = half * 16 + i;
    ks[8 + i] = half * 16 + 8 + i;
  }
  v16u16 A0, A1, B0, B1;
#pragma unroll
  for (int i = 0; i < 16; ++i) {
    A0[i] = (unsigned short)(0x3F80u | ((unsigned)(__popc(l16 & ks[i]) & 1) << 15));
    A1[i] = (unsigned short)(0x3F80u | ((unsigned)(__popc((16 + l16) & ks[i]) & 1) << 15));
    B0[i] = (unsigned short)(0x3F80u | ((unsigned)(__popc((kb + i) & l16) & 1) << 15));
    B1[i] = (unsigned short)(0x3F80u | ((unsigned)(__popc((kb + i) & (16 + l16)) & 1) << 15));
  }
  pa0 = __builtin_bit_cast(v16bf, A0);
  pa1 = __builtin_bit_cast(v16bf, A1);
  pb0 = __builtin_bit_cast(v16bf, B0);
  pb1 = __builtin_bit_cast(v16bf, B1);
}

// Length-1024 Hadamard transform (1/32 normalization) on a vector held
// COLUMN-MAJOR in LDS: element r, p=r>>5, q=r&31 at rb[q*RBS + p].
// One full wave32; EXEC all ones around each WMMA.
__device__ __forceinline__ void had1024_wave(float* rb, __bf16* st, int lane,
                                             v16bf pa0, v16bf pa1,
                                             v16bf pb0, v16bf pb1) {
  const int half = lane >> 4;
  const int l16  = lane & 15;
  const int kb   = half * 16;

  // Phase 1: T = H32 * A.  B-operand = A[kb+i][col] -> 4x ds_load_b128.
#pragma unroll
  for (int ti = 0; ti < 2; ++ti) {
#pragma unroll
    for (int tj = 0; tj < 2; ++tj) {
      const int col = tj * 16 + l16;
      const float* bp = rb + col * RBS + kb;
      const float4 f0 = *(const float4*)(bp + 0);
      const float4 f1 = *(const float4*)(bp + 4);
      const float4 f2 = *(const float4*)(bp + 8);
      const float4 f3 = *(const float4*)(bp + 12);
      v16bf b;
      b[0] = (__bf16)f0.x; b[1] = (__bf16)f0.y; b[2]  = (__bf16)f0.z; b[3]  = (__bf16)f0.w;
      b[4] = (__bf16)f1.x; b[5] = (__bf16)f1.y; b[6]  = (__bf16)f1.z; b[7]  = (__bf16)f1.w;
      b[8] = (__bf16)f2.x; b[9] = (__bf16)f2.y; b[10] = (__bf16)f2.z; b[11] = (__bf16)f2.w;
      b[12]= (__bf16)f3.x; b[13]= (__bf16)f3.y; b[14] = (__bf16)f3.z; b[15] = (__bf16)f3.w;
      v8f c = {};
      c = __builtin_amdgcn_wmma_f32_16x16x32_bf16(false, ti ? pa1 : pa0, false, b,
                                                  (short)0, c, false, false);
#pragma unroll
      for (int i = 0; i < 8; ++i)
        st[(ti * 16 + half * 8 + i) * STS + tj * 16 + l16] = (__bf16)c[i];
    }
  }

  // Phase 2: B2 = (T * H32) / 32.
#pragma unroll
  for (int ti = 0; ti < 2; ++ti) {
#pragma unroll
    for (int tj = 0; tj < 2; ++tj) {
      const int m = ti * 16 + l16;
      const int n = tj * 16 + l16;
      const v8bf alo = *(const v8bf*)(st + m * STS + half * 16);
      const v8bf ahi = *(const v8bf*)(st + m * STS + half * 16 + 8);
      v16bf a = __builtin_shufflevector(alo, ahi, 0, 1, 2, 3, 4, 5, 6, 7,
                                        8, 9, 10, 11, 12, 13, 14, 15);
      v8f c = {};
      c = __builtin_amdgcn_wmma_f32_16x16x32_bf16(false, a, false, tj ? pb1 : pb0,
                                                  (short)0, c, false, false);
      float* op = rb + n * RBS + ti * 16 + half * 8;
      *(float4*)(op + 0) = make_float4(c[0] * 0.03125f, c[1] * 0.03125f,
                                       c[2] * 0.03125f, c[3] * 0.03125f);
      *(float4*)(op + 4) = make_float4(c[4] * 0.03125f, c[5] * 0.03125f,
                                       c[6] * 0.03125f, c[7] * 0.03125f);
    }
  }
}

// Pass 1 / Pass 3: transform along the contiguous (low-10-bit) axis.
__global__ __launch_bounds__(128) void fwht_rows_kernel(
    const float* __restrict__ reIn, const float* __restrict__ imIn,
    float* __restrict__ out) {
  __shared__ __align__(16) float  rb[4][32 * RBS];
  __shared__ __align__(16) __bf16 st[4][32 * STS];

  const int lane = threadIdx.x & 31;
  const int wave = threadIdx.x >> 5;
  const size_t row = (size_t)blockIdx.x * 4 + wave;

  v16bf pa0, pa1, pb0, pb1;
  had_patterns(lane, pa0, pa1, pb0, pb1);

  const float* src = (row < (size_t)16384)
                         ? (reIn + row * 1024)
                         : (imIn + (row - 16384) * 1024);
  float* dst = out + row * 1024;

  const int q0    = (lane & 7) * 4;
  const int pbase = lane >> 3;
#pragma unroll
  for (int i = 0; i < 8; ++i) {
    const float4 v = ((const float4*)src)[i * 32 + lane];
    const int p = i * 4 + pbase;
    rb[wave][(q0 + 0) * RBS + p] = v.x;
    rb[wave][(q0 + 1) * RBS + p] = v.y;
    rb[wave][(q0 + 2) * RBS + p] = v.z;
    rb[wave][(q0 + 3) * RBS + p] = v.w;
  }

  had1024_wave(rb[wave], st[wave], lane, pa0, pa1, pb0, pb1);

#pragma unroll
  for (int i = 0; i < 8; ++i) {
    const int p = i * 4 + pbase;
    float4 v;
    v.x = rb[wave][(q0 + 0) * RBS + p];
    v.y = rb[wave][(q0 + 1) * RBS + p];
    v.z = rb[wave][(q0 + 2) * RBS + p];
    v.w = rb[wave][(q0 + 3) * RBS + p];
    ((float4*)dst)[i * 32 + lane] = v;
  }
}

// Pass 2: per workgroup, a 1024 x NCOL column slab (re+im) of one batch row.
// H along the high-10-bit axis, phase rotation, H along the same axis again.
// 256 threads = 8 waves, 2 columns per wave. In place on d_out.
__global__ __launch_bounds__(256) void fwht_cols_phase_kernel(
    float* __restrict__ out, const float* __restrict__ tptr) {
  __shared__ __align__(16) float  slabR[1024 * SLS];
  __shared__ __align__(16) float  slabI[1024 * SLS];
  __shared__ __align__(16) float  rb[8][32 * RBS];
  __shared__ __align__(16) __bf16 st[8][32 * STS];
  __shared__ float tc[21], ts[21];

  const int tid  = threadIdx.x;
  const int lane = tid & 31;
  const int wave = tid >> 5;
  const int b    = blockIdx.x >> 6;           // batch row (64 col-groups per row)
  const int c0   = (blockIdx.x & 63) * NCOL;  // column-group base

  float* baseR = out + (size_t)b * DIMX + c0;
  float* baseI = out + (size_t)BATCHX * DIMX + (size_t)b * DIMX + c0;

  v16bf pa0, pa1, pb0, pb1;
  had_patterns(lane, pa0, pa1, pb0, pb1);

  // 21 distinct phase values: spec = 20 - 2k, k = 0..20.
  if (tid < 21) {
    float s, c;
    __sincosf(tptr[0] * (float)(20 - 2 * tid), &s, &c);
    tc[tid] = c;
    ts[tid] = s;
  }

  // Slab load: TDM DMA with pad_enable (preferred), else vectorized copies
  // into the same padded layout.
#if defined(HAVE_TDM)
  if (tid == 0) {
    tdm_load_slab(baseR, lds_offset_of(slabR));
    tdm_load_slab(baseI, lds_offset_of(slabI));
    __builtin_amdgcn_s_wait_tensorcnt(0);
  }
#else
  for (int e = tid; e < 4096; e += 256) {
    const int r = e >> 2, c4 = (e & 3) * 4;
    *(float4*)&slabR[r * SLS + c4] = *(const float4*)(baseR + (size_t)r * 1024 + c4);
    *(float4*)&slabI[r * SLS + c4] = *(const float4*)(baseI + (size_t)r * 1024 + c4);
  }
#endif
  __syncthreads();

  // --- H along the rows-axis, first application. Wave owns columns
  // cc = 2*wave, 2*wave+1; gather col into column-major rb (8x ds_store_b128).
#pragma unroll 1
  for (int comp = 0; comp < 2; ++comp) {
    float* slab = comp ? slabI : slabR;
#pragma unroll 1
    for (int cs = 0; cs < 2; ++cs) {
      const int cc = wave * 2 + cs;
#pragma unroll
      for (int i8 = 0; i8 < 8; ++i8) {
        float4 v;
        v.x = slab[((i8 * 4 + 0) * 32 + lane) * SLS + cc];
        v.y = slab[((i8 * 4 + 1) * 32 + lane) * SLS + cc];
        v.z = slab[((i8 * 4 + 2) * 32 + lane) * SLS + cc];
        v.w = slab[((i8 * 4 + 3) * 32 + lane) * SLS + cc];
        *(float4*)&rb[wave][lane * RBS + i8 * 4] = v;
      }
      had1024_wave(rb[wave], st[wave], lane, pa0, pa1, pb0, pb1);
#pragma unroll
      for (int i8 = 0; i8 < 8; ++i8) {
        const float4 v = *(const float4*)&rb[wave][lane * RBS + i8 * 4];
        slab[((i8 * 4 + 0) * 32 + lane) * SLS + cc] = v.x;
        slab[((i8 * 4 + 1) * 32 + lane) * SLS + cc] = v.y;
        slab[((i8 * 4 + 2) * 32 + lane) * SLS + cc] = v.z;
        slab[((i8 * 4 + 3) * 32 + lane) * SLS + cc] = v.w;
      }
    }
  }
  __syncthreads();

  // --- Pointwise phase rotation via the 21-entry table.
  for (int e = tid; e < 1024 * NCOL; e += 256) {
    const int r  = e >> 4;
    const int cc = e & 15;
    const int idx = r * SLS + cc;
    const int pc = __popc(r) + __popc(c0 + cc);
    const float c = tc[pc], s = ts[pc];
    const float xr = slabR[idx], xi = slabI[idx];
    slabR[idx] = c * xr + s * xi;
    slabI[idx] = c * xi - s * xr;
  }
  __syncthreads();

  // --- H along the rows-axis, second application (start of second FWHT).
#pragma unroll 1
  for (int comp = 0; comp < 2; ++comp) {
    float* slab = comp ? slabI : slabR;
#pragma unroll 1
    for (int cs = 0; cs < 2; ++cs) {
      const int cc = wave * 2 + cs;
#pragma unroll
      for (int i8 = 0; i8 < 8; ++i8) {
        float4 v;
        v.x = slab[((i8 * 4 + 0) * 32 + lane) * SLS + cc];
        v.y = slab[((i8 * 4 + 1) * 32 + lane) * SLS + cc];
        v.z = slab[((i8 * 4 + 2) * 32 + lane) * SLS + cc];
        v.w = slab[((i8 * 4 + 3) * 32 + lane) * SLS + cc];
        *(float4*)&rb[wave][lane * RBS + i8 * 4] = v;
      }
      had1024_wave(rb[wave], st[wave], lane, pa0, pa1, pb0, pb1);
#pragma unroll
      for (int i8 = 0; i8 < 8; ++i8) {
        const float4 v = *(const float4*)&rb[wave][lane * RBS + i8 * 4];
        slab[((i8 * 4 + 0) * 32 + lane) * SLS + cc] = v.x;
        slab[((i8 * 4 + 1) * 32 + lane) * SLS + cc] = v.y;
        slab[((i8 * 4 + 2) * 32 + lane) * SLS + cc] = v.z;
        slab[((i8 * 4 + 3) * 32 + lane) * SLS + cc] = v.w;
      }
    }
  }
  __syncthreads();

  // Writeback from the padded layout: TDM padding is load-only, so use async
  // LDS->global b128 stores (aligned: row base 80B, chunks 16B).
#if defined(ASYNC_STORE)
  for (int e = tid; e < 4096; e += 256) {
    const int r = e >> 2, c4 = (e & 3) * 4;
    __builtin_amdgcn_global_store_async_from_lds_b128(
        ASYNC_GPTR(baseR + (size_t)r * 1024 + c4), ASYNC_LPTR(&slabR[r * SLS + c4]), 0, 0);
    __builtin_amdgcn_global_store_async_from_lds_b128(
        ASYNC_GPTR(baseI + (size_t)r * 1024 + c4), ASYNC_LPTR(&slabI[r * SLS + c4]), 0, 0);
  }
  wait_async0();
#else
  for (int e = tid; e < 4096; e += 256) {
    const int r = e >> 2, c4 = (e & 3) * 4;
    *(float4*)(baseR + (size_t)r * 1024 + c4) = *(const float4*)&slabR[r * SLS + c4];
    *(float4*)(baseI + (size_t)r * 1024 + c4) = *(const float4*)&slabI[r * SLS + c4];
  }
#endif
}

extern "C" void kernel_launch(void* const* d_in, const int* in_sizes, int n_in,
                              void* d_out, int out_size, void* d_ws, size_t ws_size,
                              hipStream_t stream) {
  const float* xr = (const float*)d_in[0];
  const float* xi = (const float*)d_in[1];
  const float* t  = (const float*)d_in[2];
  float* out = (float*)d_out;  // 2 * 16 * 2^20 f32 (re block then im block)

  (void)d_ws; (void)ws_size; (void)in_sizes; (void)n_in; (void)out_size;

  // Pass 1: low-axis H per component -> d_out (working set then lives in L2).
  fwht_rows_kernel<<<8192, 128, 0, stream>>>(xr, xi, out);
  // Pass 2: high-axis H, phase, high-axis H; in place on d_out.
  fwht_cols_phase_kernel<<<1024, 256, 0, stream>>>(out, t);
  // Pass 3: final low-axis H; in place on d_out.
  fwht_rows_kernel<<<8192, 128, 0, stream>>>(out, out + (size_t)BATCHX * DIMX, out);
}
